// Head_4355096838316
// MI455X (gfx1250) — matrix operand order repaired and checked by reading, not compiled, not run on hardware
//
#include <hip/hip_runtime.h>

#define Bsz 8
#define Tsz 2048
#define Csz 1024
#define Hsz 64

typedef __attribute__((ext_vector_type(16))) _Float16     v16h;
typedef __attribute__((ext_vector_type(8)))  _Float16     v8h;
typedef __attribute__((ext_vector_type(8)))  float        v8f;
typedef __attribute__((ext_vector_type(4)))  unsigned int v4u;
typedef __attribute__((ext_vector_type(8)))  int          v8i;
typedef __attribute__((ext_vector_type(4)))  int          v4i;

#if defined(__has_builtin)
#if __has_builtin(__builtin_amdgcn_tensor_load_to_lds) && \
    __has_builtin(__builtin_amdgcn_s_wait_tensorcnt)
#define USE_TDM 1
#endif
#endif
#ifndef USE_TDM
#define USE_TDM 0
#endif

__device__ __forceinline__ v8f wmma_f16(v16h a, v16h b, v8f c) {
  // D = A(16x32 f16) * B(32x16 f16) + C(16x16 f32)
  return __builtin_amdgcn_wmma_f32_16x16x32_f16(
      /*neg_a=*/false, a, /*neg_b=*/false, b,
      /*c_mod=*/(short)0, c, /*reuse_a=*/false, /*reuse_b=*/false);
}

#if USE_TDM
// Issue a 2-D TDM tile load: rows x row_elems f16 elements, row pitch
// stride0 (f16 units) in global, padded in LDS per pad codes.
// D# per CDNA5 ISA §8.3/8.4 (group0: count/lds/global/type; group1: sizes).
__device__ __forceinline__ void tdm_load_2d_f16(
    unsigned lds_byte_off, const void* gaddr, unsigned row_elems,
    unsigned rows, unsigned tensor_d0, unsigned tensor_d1, unsigned stride0,
    unsigned pad_interval_code, unsigned pad_amount_code) {
  const unsigned long long ga = (unsigned long long)gaddr;
  v4u g0;
  g0[0] = 1u;                                   // [1:0] count=1 (valid D#)
  g0[1] = lds_byte_off;                         // [63:32] lds_addr
  g0[2] = (unsigned)(ga & 0xFFFFFFFFu);         // [95:64] global_addr lo
  g0[3] = (unsigned)((ga >> 32) & 0x01FFFFFFu)  // [120:96] global_addr hi
          | 0x80000000u;                        // [127:126] type=2 (image)
  v8i g1;
  g1[0] = (int)((1u << 16)                      // data_size=1 -> 2 bytes
                | (1u << 20)                    // pad_enable
                | (pad_interval_code << 22) | (pad_amount_code << 25));
  g1[1] = (int)(tensor_d0 << 16);               // tensor_dim0 lo16 @ [63:48]
  g1[2] = (int)(((tensor_d0 >> 16) & 0xFFFFu)   // tensor_dim0 hi16 @ [79:64]
                | (tensor_d1 << 16));           // tensor_dim1 lo16 @ [95:80]
  g1[3] = (int)(((tensor_d1 >> 16) & 0xFFFFu)   // tensor_dim1 hi16 @ [111:96]
                | (row_elems << 16));           // tile_dim0 @ [127:112]
  g1[4] = (int)(rows & 0xFFFFu);                // tile_dim1; tile_dim2=0
  g1[5] = (int)stride0;                         // tensor_dim0_stride lo32
  g1[6] = 0;                                    // stride0 hi16 | dim1_stride
  g1[7] = 0;
  const v4i z4 = {0, 0, 0, 0};
#if __clang_major__ >= 23
  const v8i z8 = {0, 0, 0, 0, 0, 0, 0, 0};
  __builtin_amdgcn_tensor_load_to_lds(g0, g1, z4, z4, z8, 0);
#else
  __builtin_amdgcn_tensor_load_to_lds(g0, g1, z4, z4, 0);
#endif
}

__device__ __forceinline__ unsigned lds_offset(const void* p) {
  // Flat shared-aperture addresses carry the LDS byte offset in bits [31:0].
  return (unsigned)(unsigned long long)p;
}
#endif  // USE_TDM

// ---------------------------------------------------------------------------
// Kernel 1: transpose + f16-convert the three weight matrices [C,H] -> [H,C]
// so that WMMA B-fragments (K-contiguous per lane) load contiguously.
// ---------------------------------------------------------------------------
__global__ __launch_bounds__(256) void prep_w_kernel(
    const float* __restrict__ wq, const float* __restrict__ wk,
    const float* __restrict__ wv,
    _Float16* __restrict__ wTq, _Float16* __restrict__ wTk,
    _Float16* __restrict__ wTv) {
  const int idx = blockIdx.x * 256 + threadIdx.x;   // 0 .. H*C-1
  const int h = idx >> 10;                          // / C
  const int c = idx & (Csz - 1);                    // % C
  const size_t src = (size_t)c * Hsz + h;
  const size_t dst = (size_t)h * Csz + c;
  wTq[dst] = (_Float16)wq[src];
  wTk[dst] = (_Float16)wk[src];
  wTv[dst] = (_Float16)wv[src];
}

// ---------------------------------------------------------------------------
// Kernel 2: projections q,k,v = x @ w via WMMA.
// One block = 16 rows of one batch; 4 waves, wave w owns H columns [16w,16w+16).
// q,k stored row-major [B,T,H] f16; v stored transposed [B,H,T] f16.
// ---------------------------------------------------------------------------
__global__ __launch_bounds__(128) void proj_kernel(
    const float* __restrict__ x,
    const _Float16* __restrict__ wTq, const _Float16* __restrict__ wTk,
    const _Float16* __restrict__ wTv,
    _Float16* __restrict__ qm, _Float16* __restrict__ km,
    _Float16* __restrict__ vTm) {
  const int lane = threadIdx.x & 31;
  const int wave = threadIdx.x >> 5;
  const int mrow = lane & 15;        // A-layout M / C-layout N
  const int half = lane >> 4;
  const int c0   = half << 3;        // unused; A-frag K chunks start at kcol, kcol+8
  const int kcol = half << 4;        // B-frag K base (0 or 16)
  const int bidx = blockIdx.x >> 7;
  const int t0   = (blockIdx.x & 127) << 4;

  const float*    xrow = x   + ((size_t)bidx * Tsz + t0 + mrow) * Csz;
  const _Float16* wqr  = wTq + (size_t)(wave * 16 + mrow) * Csz + kcol;
  const _Float16* wkr  = wTk + (size_t)(wave * 16 + mrow) * Csz + kcol;
  const _Float16* wvr  = wTv + (size_t)(wave * 16 + mrow) * Csz + kcol;

  v8f aq, ak, av;
#pragma unroll
  for (int r = 0; r < 8; ++r) { aq[r] = 0.f; ak[r] = 0.f; av[r] = 0.f; }

  for (int kk = 0; kk < Csz; kk += 32) {
    v8f x0 = *(const v8f*)(xrow + kk + kcol);
    v8f x1 = *(const v8f*)(xrow + kk + kcol + 8);
    v16h a;
#pragma unroll
    for (int i = 0; i < 8; ++i) {
      a[i]     = (_Float16)x0[i];
      a[i + 8] = (_Float16)x1[i];
    }
    v16h bq = *(const v16h*)(wqr + kk);
    v16h bk = *(const v16h*)(wkr + kk);
    v16h bv = *(const v16h*)(wvr + kk);
    aq = wmma_f16(a, bq, aq);
    ak = wmma_f16(a, bk, ak);
    av = wmma_f16(a, bv, av);
  }

  const int h = wave * 16 + mrow;
#pragma unroll
  for (int r = 0; r < 8; ++r) {
    const int row = t0 + r + (half << 3);
    const size_t base = ((size_t)bidx * Tsz + row) * Hsz + h;
    qm[base] = (_Float16)aq[r];
    km[base] = (_Float16)ak[r];
    vTm[((size_t)bidx * Hsz + h) * Tsz + row] = (_Float16)av[r];
  }
}

// ---------------------------------------------------------------------------
// Kernel 3: flash attention, one wave per 16-query tile, 32-key blocks.
// TDM double-buffers the k tile (32x64 -> LDS rows padded to 80 f16) and the
// vT tile (64x32 -> LDS rows padded to 48 f16) while the wave runs WMMAs.
// ---------------------------------------------------------------------------
#define KPITCH 80   // 64 data + 16 pad f16 (pad_interval=32dw, pad=8dw)
#define VPITCH 48   // 32 data + 16 pad f16 (pad_interval=16dw, pad=8dw)

__global__ __launch_bounds__(32) void attn_kernel(
    const _Float16* __restrict__ qm, const _Float16* __restrict__ km,
    const _Float16* __restrict__ vTm, float* __restrict__ out) {
  __shared__ _Float16 Plds[16 * 32];
#if USE_TDM
  __shared__ _Float16 kbuf[2][32 * KPITCH];
  __shared__ _Float16 vbuf[2][64 * VPITCH];
#endif

  const int lane = threadIdx.x & 31;
  const int mrow = lane & 15;
  const int half = lane >> 4;
  const int c0   = half << 3;
  const int kcol = half << 4;
  const int bidx = blockIdx.x >> 7;
  const int q0   = (blockIdx.x & 127) << 4;

  // Pre-scaled q A-fragments (scale = H^-0.5 = 0.125, exact in f16).
  v16h qa[2];
  {
    const _Float16* qrow = qm + ((size_t)bidx * Tsz + q0 + mrow) * Hsz;
#pragma unroll
    for (int kk = 0; kk < 2; ++kk) {
      v8h a0 = *(const v8h*)(qrow + kk * 32 + kcol);
      v8h a1 = *(const v8h*)(qrow + kk * 32 + kcol + 8);
#pragma unroll
      for (int i = 0; i < 8; ++i) {
        qa[kk][i]     = a0[i] * (_Float16)0.125f;
        qa[kk][i + 8] = a1[i] * (_Float16)0.125f;
      }
    }
  }

  v8f oacc[4];
#pragma unroll
  for (int n = 0; n < 4; ++n)
#pragma unroll
    for (int r = 0; r < 8; ++r) oacc[n][r] = 0.f;

  float mrun[8], lrun[8];
#pragma unroll
  for (int r = 0; r < 8; ++r) { mrun[r] = -3.0e38f; lrun[r] = 0.f; }

  const int kbmax = (q0 + 15) >> 5;

#if USE_TDM
  // Prologue: stage key block 0 into buffer 0 via the Tensor Data Mover.
  tdm_load_2d_f16(lds_offset(&kbuf[0][0]),
                  km + ((size_t)bidx * Tsz) * Hsz,
                  /*row_elems=*/64, /*rows=*/32, /*t_d0=*/64, /*t_d1=*/Tsz,
                  /*stride0=*/64, /*interval=32dw*/ 4, /*pad=8dw*/ 7);
  tdm_load_2d_f16(lds_offset(&vbuf[0][0]),
                  vTm + (size_t)bidx * Hsz * Tsz,
                  /*row_elems=*/32, /*rows=*/64, /*t_d0=*/Tsz, /*t_d1=*/Hsz,
                  /*stride0=*/Tsz, /*interval=16dw*/ 3, /*pad=8dw*/ 7);
#endif

  for (int kb = 0; kb <= kbmax; ++kb) {
    const int k0 = kb << 5;

#if USE_TDM
    if (kb < kbmax) {  // wave-uniform: prefetch next key block, then make
      const int kn = k0 + 32;  // sure block kb (2 older TDM ops) has landed.
      tdm_load_2d_f16(lds_offset(&kbuf[(kb + 1) & 1][0]),
                      km + ((size_t)bidx * Tsz + kn) * Hsz,
                      64, 32, 64, Tsz, 64, 4, 7);
      tdm_load_2d_f16(lds_offset(&vbuf[(kb + 1) & 1][0]),
                      vTm + (size_t)bidx * Hsz * Tsz + kn,
                      32, 64, Tsz, Hsz, Tsz, 3, 7);
      __builtin_amdgcn_s_wait_tensorcnt((short)2);
    } else {
      __builtin_amdgcn_s_wait_tensorcnt((short)0);
    }
    asm volatile("" ::: "memory");
    const _Float16* kbufc = &kbuf[kb & 1][0];
    const _Float16* vbufc = &vbuf[kb & 1][0];
#endif

    // ---- S = q * k^T (two 16x16 tiles over 32 keys, K=64 = 2x32) ----
    v8f S[2];
#pragma unroll
    for (int nt = 0; nt < 2; ++nt) {
#pragma unroll
      for (int r = 0; r < 8; ++r) S[nt][r] = 0.f;
#if USE_TDM
      const _Float16* krow = kbufc + (nt * 16 + mrow) * KPITCH + kcol;
#else
      const _Float16* krow =
          km + ((size_t)bidx * Tsz + k0 + nt * 16 + mrow) * Hsz + kcol;
#endif
      v16h b0 = *(const v16h*)(krow);
      v16h b1 = *(const v16h*)(krow + 32);
      S[nt] = wmma_f16(qa[0], b0, S[nt]);
      S[nt] = wmma_f16(qa[1], b1, S[nt]);
    }

#if !USE_TDM
    if (kb < kbmax)
      __builtin_prefetch(km + ((size_t)bidx * Tsz + k0 + 32 + mrow) * Hsz, 0, 1);
#endif

    // ---- causal mask + online softmax (row M = r + 8*half, col N = mrow) --
#pragma unroll
    for (int r = 0; r < 8; ++r) {
      const int qi = q0 + r + (half << 3);
      float s0 = (k0 + mrow      <= qi) ? S[0][r] : -3.0e38f;
      float s1 = (k0 + 16 + mrow <= qi) ? S[1][r] : -3.0e38f;
      float t = fmaxf(s0, s1);
      t = fmaxf(t, __shfl_xor(t, 1, 32));
      t = fmaxf(t, __shfl_xor(t, 2, 32));
      t = fmaxf(t, __shfl_xor(t, 4, 32));
      t = fmaxf(t, __shfl_xor(t, 8, 32));   // stays within the 16-lane half
      const float mnew  = fmaxf(mrun[r], t);
      const float alpha = __expf(mrun[r] - mnew);
      const float p0    = __expf(s0 - mnew);
      const float p1    = __expf(s1 - mnew);
      float rs = p0 + p1;
      rs += __shfl_xor(rs, 1, 32);
      rs += __shfl_xor(rs, 2, 32);
      rs += __shfl_xor(rs, 4, 32);
      rs += __shfl_xor(rs, 8, 32);
      lrun[r] = lrun[r] * alpha + rs;
      mrun[r] = mnew;
#pragma unroll
      for (int n = 0; n < 4; ++n) oacc[n][r] *= alpha;
      const int M = r + (half << 3);
      Plds[M * 32 + mrow]      = (_Float16)p0;
      Plds[M * 32 + 16 + mrow] = (_Float16)p1;
    }

    asm volatile("s_wait_dscnt 0x0" ::: "memory");  // LDS stores visible

    // ---- re-read P in WMMA A layout ----
    v16h pa;
    {
      const _Float16* pr = &Plds[mrow * 32 + kcol];
      v8h p0v = *(const v8h*)(pr);
      v8h p1v = *(const v8h*)(pr + 8);
#pragma unroll
      for (int i = 0; i < 8; ++i) { pa[i] = p0v[i]; pa[i + 8] = p1v[i]; }
    }

    // ---- O += P * V (vT rows are K-contiguous) ----
#pragma unroll
    for (int n = 0; n < 4; ++n) {
#if USE_TDM
      const _Float16* vrow = vbufc + (n * 16 + mrow) * VPITCH + kcol;
#else
      const _Float16* vrow =
          vTm + ((size_t)bidx * Hsz + n * 16 + mrow) * Tsz + k0 + kcol;
#endif
      v16h bv = *(const v16h*)vrow;
      oacc[n] = wmma_f16(pa, bv, oacc[n]);
    }
  }

  // ---- epilogue: divide by running sum, write fp32 output ----
#pragma unroll
  for (int r = 0; r < 8; ++r) {
    const float inv = 1.0f / lrun[r];
    const size_t row = (size_t)bidx * Tsz + q0 + r + (half << 3);
#pragma unroll
    for (int n = 0; n < 4; ++n)
      out[row * Hsz + n * 16 + mrow] = oacc[n][r] * inv;
  }
}

// ---------------------------------------------------------------------------
extern "C" void kernel_launch(void* const* d_in, const int* in_sizes, int n_in,
                              void* d_out, int out_size, void* d_ws,
                              size_t ws_size, hipStream_t stream) {
  (void)in_sizes; (void)n_in; (void)out_size; (void)ws_size;
  const float* x  = (const float*)d_in[0];
  const float* wq = (const float*)d_in[1];
  const float* wk = (const float*)d_in[2];
  const float* wv = (const float*)d_in[3];
  float* out = (float*)d_out;

  char* ws = (char*)d_ws;
  constexpr size_t WT_BYTES  = (size_t)Hsz * Csz * 2;        // 128 KB each
  constexpr size_t QKV_BYTES = (size_t)Bsz * Tsz * Hsz * 2;  // 2 MB each
  _Float16* wTq = (_Float16*)(ws);
  _Float16* wTk = (_Float16*)(ws + WT_BYTES);
  _Float16* wTv = (_Float16*)(ws + 2 * WT_BYTES);
  _Float16* qm  = (_Float16*)(ws + 3 * WT_BYTES);
  _Float16* km  = (_Float16*)(ws + 3 * WT_BYTES + QKV_BYTES);
  _Float16* vTm = (_Float16*)(ws + 3 * WT_BYTES + 2 * QKV_BYTES);

  prep_w_kernel<<<(Hsz * Csz) / 256, 256, 0, stream>>>(wq, wk, wv,
                                                       wTq, wTk, wTv);
  proj_kernel<<<Bsz * (Tsz / 16), 128, 0, stream>>>(x, wTq, wTk, wTv,
                                                    qm, km, vTm);
  attn_kernel<<<Bsz * (Tsz / 16), 32, 0, stream>>>(qm, km, vTm, out);
}
